// MultiLayerGNN_8710193676618
// MI455X (gfx1250) — matrix operand order repaired and checked by reading, not compiled, not run on hardware
//
#include <hip/hip_runtime.h>

#define BN 8
#define NN 2048
#define FF 512
#define HH 256
#define LL 3

typedef __attribute__((ext_vector_type(16))) _Float16 v16h;
typedef __attribute__((ext_vector_type(8)))  _Float16 v8h;
typedef __attribute__((ext_vector_type(8)))  float    v8f;

// ---------------------------------------------------------------------------
// A-fragment (16x32 f16, M x K) load, contiguous K order (matches B):
//   lanes 0-15 : M=lane,    halves 0-15 -> K 0-15
//   lanes 16-31: M=lane-16, halves 0-15 -> K 16-31
// p already points at row base + (hiHalf ? 16 : 0); two contiguous 16B loads.
// ---------------------------------------------------------------------------
__device__ __forceinline__ v16h load_a_frag(const _Float16* p) {
    union { v16h v; v8h h[2]; } u;
    u.h[0] = *(const v8h*)(p);
    u.h[1] = *(const v8h*)(p + 8);
    return u.v;
}

// ---------------------------------------------------------------------------
// GEMM: C[M x 256] = A[M x K] (row-major f16) * B[K x 256], where B is stored
// transposed as BT[256 x K] row-major f16 so the 32x16 B fragment is one
// contiguous 32B load per lane (lanes 0-15: K 0-15, lanes 16-31: K 16-31).
// Block = 8 waves, tile 64(M) x 256(N); wave = 2 M-subtiles x 4 N-subtiles.
// K loop unrolled by 2 with ping-pong fragment buffers (X/Y): consume X,
// refill X at k+64; consume Y, refill Y at k+96. No register rotation copies,
// 12 b128 loads in flight under every 8-WMMA burst. Requires K % 64 == 0
// (holds for K = 2048 / 512 / 256 here).
// A stream (the only HBM-resident operand) gets global_prefetch ~1KB ahead.
// ---------------------------------------------------------------------------
template <bool OUT_HALF>
__global__ __launch_bounds__(256) void gemm256_wmma(
    const _Float16* __restrict__ A, const _Float16* __restrict__ BT,
    void* __restrict__ Cout, int K,
    long aBatch, long bBatch, long cBatch)
{
    const int lane   = threadIdx.x & 31;
    const int wave   = threadIdx.x >> 5;
    const int wm     = wave & 1;    // 0..1 -> 32-row group
    const int wn     = wave >> 1;   // 0..3 -> 64-col group
    const int mlane  = lane & 15;
    const int hiHalf = (lane >= 16);

    const long b = blockIdx.z;
    A  += b * aBatch;
    BT += b * bBatch;

    const int rowBlk = blockIdx.x * 64 + wm * 32;
    const int colBlk = wn * 64;

    const _Float16* pa0 = A  + (long)(rowBlk + mlane) * K + (hiHalf ? 16 : 0);
    const _Float16* pa1 = pa0 + (long)16 * K;
    const _Float16* pb0 = BT + (long)(colBlk + mlane) * K + (hiHalf ? 16 : 0);

    v8f acc[2][4] = {};

    v16h xa0, xa1, xb[4];   // ping buffer
    v16h ya0, ya1, yb[4];   // pong buffer

    auto loadX = [&](int k) {
        xa0 = load_a_frag(pa0 + k);
        xa1 = load_a_frag(pa1 + k);
        #pragma unroll
        for (int j = 0; j < 4; ++j)
            xb[j] = *(const v16h*)(pb0 + (long)(j * 16) * K + k);
    };
    auto loadY = [&](int k) {
        ya0 = load_a_frag(pa0 + k);
        ya1 = load_a_frag(pa1 + k);
        #pragma unroll
        for (int j = 0; j < 4; ++j)
            yb[j] = *(const v16h*)(pb0 + (long)(j * 16) * K + k);
    };
    auto mma = [&](const v16h& a0, const v16h& a1, const v16h* bf) {
        #pragma unroll
        for (int j = 0; j < 4; ++j) {
            acc[0][j] = __builtin_amdgcn_wmma_f32_16x16x32_f16(
                false, a0, false, bf[j], (short)0, acc[0][j], false, false);
            acc[1][j] = __builtin_amdgcn_wmma_f32_16x16x32_f16(
                false, a1, false, bf[j], (short)0, acc[1][j], false, false);
        }
    };

    loadX(0);
    loadY(32);
    for (int k0 = 64; k0 < K; k0 += 64) {
        // keep the HBM stream (A) ahead of the consume point
        __builtin_prefetch(pa0 + k0 + 448, 0, 1);
        __builtin_prefetch(pa1 + k0 + 448, 0, 1);
        mma(xa0, xa1, xb);
        loadX(k0);
        mma(ya0, ya1, yb);
        loadY(k0 + 32);
    }
    mma(xa0, xa1, xb);
    mma(ya0, ya1, yb);

    // C/D layout: VGPR v -> row = v + (lane>=16 ? 8 : 0), col = lane&15
    const int rOff = hiHalf ? 8 : 0;
    #pragma unroll
    for (int i = 0; i < 2; ++i) {
        #pragma unroll
        for (int j = 0; j < 4; ++j) {
            const int r0 = rowBlk + i * 16 + rOff;
            const int c  = colBlk + j * 16 + mlane;
            #pragma unroll
            for (int v = 0; v < 8; ++v) {
                const long idx = b * cBatch + (long)(r0 + v) * 256 + c;
                if (OUT_HALF) ((_Float16*)Cout)[idx] = (_Float16)acc[i][j][v];
                else          ((float*)Cout)[idx]    = acc[i][j][v];
            }
        }
    }
}

// ---------------------------------------------------------------------------
// norm_adj row: w = min(adj + I, 1) * ew ; row-normalize ; store f16
// one block per (batch,row); 2048 elements, 8 per thread
// ---------------------------------------------------------------------------
__global__ __launch_bounds__(256) void build_norm_adj(
    const float* __restrict__ adj, const float* __restrict__ ew,
    _Float16* __restrict__ nadj)
{
    const int t   = threadIdx.x;
    const int row = blockIdx.x;
    const long base = ((long)blockIdx.y * NN + row) * NN;

    float v[8];
    float s = 0.f;
    #pragma unroll
    for (int i = 0; i < 8; ++i) {
        const int c = i * 256 + t;
        float a = adj[base + c] + (c == row ? 1.f : 0.f);
        a = fminf(a, 1.f);
        const float w = a * ew[base + c];
        v[i] = w;
        s += w;
    }
    __shared__ float red[256];
    red[t] = s; __syncthreads();
    for (int st = 128; st > 0; st >>= 1) {
        if (t < st) red[t] += red[t + st];
        __syncthreads();
    }
    const float inv = 1.f / (red[0] + 1e-8f);
    #pragma unroll
    for (int i = 0; i < 8; ++i)
        nadj[base + i * 256 + t] = (_Float16)(v[i] * inv);
}

// ---------------------------------------------------------------------------
// elementwise f32 -> f16
// ---------------------------------------------------------------------------
__global__ void cvt_to_f16(const float* __restrict__ in, _Float16* __restrict__ out, long n)
{
    long i = (long)blockIdx.x * blockDim.x + threadIdx.x;
    const long stride = (long)gridDim.x * blockDim.x;
    for (; i < n; i += stride) out[i] = (_Float16)in[i];
}

// ---------------------------------------------------------------------------
// 32x32 LDS transpose, f32 in -> f16 out. in: [rows x cols], out: [cols x rows]
// grid: (cols/32, rows/32, batches), block (32,8)
// ---------------------------------------------------------------------------
__global__ __launch_bounds__(256) void transpose_cvt_f16(
    const float* __restrict__ in, _Float16* __restrict__ out,
    int rows, int cols, long inB, long outB)
{
    __shared__ _Float16 tile[32][33];
    const int tx = threadIdx.x;
    const int ty = threadIdx.y;
    const long b = blockIdx.z;

    const int x = blockIdx.x * 32 + tx;          // input col
    #pragma unroll
    for (int r = 0; r < 4; ++r) {
        const int y = blockIdx.y * 32 + ty + r * 8;   // input row
        tile[ty + r * 8][tx] = (_Float16)in[b * inB + (long)y * cols + x];
    }
    __syncthreads();
    const int ox = blockIdx.y * 32 + tx;         // output col (= input row)
    #pragma unroll
    for (int r = 0; r < 4; ++r) {
        const int oy = blockIdx.x * 32 + ty + r * 8;  // output row (= input col)
        out[b * outB + (long)oy * rows + ox] = tile[tx][ty + r * 8];
    }
}

// ---------------------------------------------------------------------------
// per-row LayerNorm(+bias) -> ReLU -> optional residual. one block per row.
// ---------------------------------------------------------------------------
__global__ __launch_bounds__(256) void ln_relu(
    const float* __restrict__ lin, const float* __restrict__ bias,
    const float* __restrict__ gamma, const float* __restrict__ beta,
    const float* hres, float* hout)
{
    const int  t   = threadIdx.x;
    const long row = blockIdx.x;
    float x = lin[row * HH + t] + bias[t];

    __shared__ float red[256];
    red[t] = x; __syncthreads();
    for (int s = 128; s > 0; s >>= 1) {
        if (t < s) red[t] += red[t + s];
        __syncthreads();
    }
    const float mean = red[0] * (1.f / HH);
    __syncthreads();
    const float d = x - mean;
    red[t] = d * d; __syncthreads();
    for (int s = 128; s > 0; s >>= 1) {
        if (t < s) red[t] += red[t + s];
        __syncthreads();
    }
    const float var = red[0] * (1.f / HH);
    float y = d * rsqrtf(var + 1e-5f) * gamma[t] + beta[t];
    y = fmaxf(y, 0.f);
    if (hres) y += hres[row * HH + t];
    hout[row * HH + t] = y;
}

// ---------------------------------------------------------------------------
extern "C" void kernel_launch(void* const* d_in, const int* in_sizes, int n_in,
                              void* d_out, int out_size, void* d_ws, size_t ws_size,
                              hipStream_t stream)
{
    (void)in_sizes; (void)n_in; (void)out_size; (void)ws_size;

    const float* node_feat = (const float*)d_in[0];   // [B,N,F]
    const float* edge_w    = (const float*)d_in[1];   // [B,N,N]
    const float* adj       = (const float*)d_in[2];   // [B,N,N]
    const float* W_in      = (const float*)d_in[3];   // [F,H]
    const float* b_in      = (const float*)d_in[4];   // [H]
    const float* g_in      = (const float*)d_in[5];   // [H]
    const float* bt_in     = (const float*)d_in[6];   // [H]
    const float* W         = (const float*)d_in[7];   // [L,H,H]
    const float* bL        = (const float*)d_in[8];   // [L,H]
    const float* gammaL    = (const float*)d_in[9];   // [L,H]
    const float* betaL     = (const float*)d_in[10];  // [L,H]
    float* hout = (float*)d_out;                      // running h [B,N,H] f32

    char* ws = (char*)d_ws;
    _Float16* nadj  = (_Float16*)(ws);              //  67,108,864 B  [B][N][N] f16
    _Float16* hT    = (_Float16*)(ws +  67108864);  //   8,388,608 B  [B][H][N] f16
    _Float16* agg16 = (_Float16*)(ws +  75497472);  //   8,388,608 B  [B*N][H] f16
    float*    lin   = (float*)   (ws +  83886080);  //  16,777,216 B  [B*N][H] f32
    _Float16* nf16  = (_Float16*)(ws + 100663296);  //  16,777,216 B  [B*N][F] f16
    _Float16* WinT  = (_Float16*)(ws + 117440512);  //     262,144 B  [H][F] f16
    _Float16* WT    = (_Float16*)(ws + 117702656);  //     393,216 B  [L][H][H] f16

    // 1. row-normalized weighted adjacency (f16, row-major = WMMA A operand)
    build_norm_adj<<<dim3(NN, BN), 256, 0, stream>>>(adj, edge_w, nadj);

    // 2. node_feat -> f16 (A operand of input projection)
    cvt_to_f16<<<4096, 256, 0, stream>>>(node_feat, nf16, (long)BN * NN * FF);

    // 3. weight transposes (B operands want K-contiguous rows)
    transpose_cvt_f16<<<dim3(HH / 32, FF / 32, 1), dim3(32, 8), 0, stream>>>(
        W_in, WinT, FF, HH, 0, 0);
    transpose_cvt_f16<<<dim3(HH / 32, HH / 32, LL), dim3(32, 8), 0, stream>>>(
        W, WT, HH, HH, (long)HH * HH, (long)HH * HH);

    // 4. input projection: lin = node_feat @ W_in   (M=16384, K=512)
    gemm256_wmma<false><<<dim3((BN * NN) / 64, 1, 1), 256, 0, stream>>>(
        nf16, WinT, lin, FF, 0, 0, 0);

    // 5. h0 = relu(LN(lin + b_in))
    ln_relu<<<BN * NN, 256, 0, stream>>>(lin, b_in, g_in, bt_in, nullptr, hout);

    // 6. hT for aggregation B operand
    transpose_cvt_f16<<<dim3(HH / 32, NN / 32, BN), dim3(32, 8), 0, stream>>>(
        hout, hT, NN, HH, (long)NN * HH, (long)NN * HH);

    for (int l = 0; l < LL; ++l) {
        // agg = norm_adj @ h   (per batch: 2048x2048 * 2048x256, f16 out)
        gemm256_wmma<true><<<dim3(NN / 64, 1, BN), 256, 0, stream>>>(
            nadj, hT, agg16, NN, (long)NN * NN, (long)HH * NN, (long)NN * HH);

        // lin = agg @ W[l]     (M=16384, K=256)
        gemm256_wmma<false><<<dim3((BN * NN) / 64, 1, 1), 256, 0, stream>>>(
            agg16, WT + (long)l * HH * HH, lin, HH, 0, 0, 0);

        // h = relu(LN(lin + b[l])) + h
        ln_relu<<<BN * NN, 256, 0, stream>>>(
            lin, bL + l * HH, gammaL + l * HH, betaL + l * HH, hout, hout);

        if (l < LL - 1)
            transpose_cvt_f16<<<dim3(HH / 32, NN / 32, BN), dim3(32, 8), 0, stream>>>(
                hout, hT, NN, HH, (long)NN * HH, (long)NN * HH);
    }
}
